// TransformerBlock_20710332301805
// MI455X (gfx1250) — matrix operand order repaired and checked by reading, not compiled, not run on hardware
//
#include <hip/hip_runtime.h>
#include <hip/hip_bf16.h>

// ---------------------------------------------------------------------------
// Types for CDNA5 WMMA bf16 (16x16x32, f32 accumulate)
// ---------------------------------------------------------------------------
typedef __attribute__((ext_vector_type(16))) __bf16 v16bf;
typedef __attribute__((ext_vector_type(8)))  __bf16 v8bf;
typedef __attribute__((ext_vector_type(8)))  float  v8f;

union FragAB { v16bf v; v8bf h[2]; };

__device__ inline v8f wmma_bf16(const FragAB& a, const FragAB& b, v8f c) {
    // (neg_a, A, neg_b, B, c_mod, C, reuse_a, reuse_b)
    return __builtin_amdgcn_wmma_f32_16x16x32_bf16(false, a.v, false, b.v,
                                                   (short)0, c, false, false);
}

// Cross-lane xor within 16-lane row groups on the VALU (v_permlane16_b32).
// Selector nibbles: sel[i] = i ^ mask, packed into two SGPR dwords.
__device__ inline float lane16_xor(float v, unsigned lo, unsigned hi) {
    unsigned s = __builtin_bit_cast(unsigned, v);
    unsigned r = __builtin_amdgcn_permlane16(s, s, lo, hi, false, false);
    return __builtin_bit_cast(float, r);
}

#define DIM      1024
#define NHEADS   16
#define HDIM     64
#define SEQ      2048
#define BATCH    4
#define MTOK     (BATCH * SEQ)     // 8192 token rows
#define HIDDEN   4096

// ---------------------------------------------------------------------------
// fp32 -> bf16 weight conversion
// ---------------------------------------------------------------------------
__global__ void cvt_bf16_kernel(const float* __restrict__ in,
                                __bf16* __restrict__ out, int n) {
    int i = blockIdx.x * 256 + threadIdx.x;
    if (i < n) out[i] = (__bf16)in[i];
}

// ---------------------------------------------------------------------------
// LayerNorm over DIM=1024, one block per token, bf16 output
// ---------------------------------------------------------------------------
__global__ __launch_bounds__(256)
void ln_kernel(const float* __restrict__ x, const float* __restrict__ g,
               const float* __restrict__ bta, __bf16* __restrict__ out) {
    __shared__ float red[2][8];
    const size_t t = blockIdx.x;
    const float* xp = x + t * DIM;
    float vals[4];
    float s = 0.f, ss = 0.f;
#pragma unroll
    for (int i = 0; i < 4; ++i) {
        float v = xp[threadIdx.x + i * 256];
        vals[i] = v; s += v; ss += v * v;
    }
#pragma unroll
    for (int m = 16; m >= 1; m >>= 1) {
        s  += __shfl_xor(s,  m, 32);
        ss += __shfl_xor(ss, m, 32);
    }
    int wv = threadIdx.x >> 5;
    if ((threadIdx.x & 31) == 0) { red[0][wv] = s; red[1][wv] = ss; }
    __syncthreads();
    float ts = 0.f, tss = 0.f;
#pragma unroll
    for (int i = 0; i < 8; ++i) { ts += red[0][i]; tss += red[1][i]; }
    const float mean = ts * (1.f / DIM);
    const float var  = tss * (1.f / DIM) - mean * mean;
    const float rstd = rsqrtf(var + 1e-5f);
#pragma unroll
    for (int i = 0; i < 4; ++i) {
        int c = threadIdx.x + i * 256;
        out[t * DIM + c] = (__bf16)((vals[i] - mean) * rstd * g[c] + bta[c]);
    }
}

// ---------------------------------------------------------------------------
// WMMA GEMM: out[M,N] = A[M,K](bf16) @ W[N,K]^T (bf16) + bias. Epilogue EPI:
//   0: f32 out   1: f32 out + residual   2: bf16 out   3: bf16 out + exact GELU
// Block = 128 threads (4 waves, 2x2). Per wave: 64(M) x 32(N), K-step 32:
//   12 global_load_b128 -> 8 v_wmma (4 A frags x 2 B frags).
// Grid: (N/64, M/128)
// ---------------------------------------------------------------------------
template <int EPI>
__global__ __launch_bounds__(128)
void gemm_kernel(const __bf16* __restrict__ A, const __bf16* __restrict__ W,
                 const float* __restrict__ bias, const float* __restrict__ residual,
                 void* __restrict__ outp, int M, int N, int K) {
    const int lane = threadIdx.x & 31;
    const int wave = threadIdx.x >> 5;
    const int ln   = lane & 15;
    const int half = lane >> 4;
    const int mBase = blockIdx.y * 128 + (wave >> 1) * 64;
    const int nBase = blockIdx.x * 64 + (wave & 1) * 32;

    v8f acc[4][2] = {};
    const __bf16* wrow0 = W + (size_t)(nBase + ln) * K;
    const __bf16* wrow1 = W + (size_t)(nBase + 16 + ln) * K;

    for (int kb = 0; kb < K; kb += 32) {
        // B fragments (32x16): lane ln = column n, elems e -> k = half*16 + e
        FragAB bf[2];
        bf[0].h[0] = *(const v8bf*)(wrow0 + kb + half * 16);
        bf[0].h[1] = *(const v8bf*)(wrow0 + kb + half * 16 + 8);
        bf[1].h[0] = *(const v8bf*)(wrow1 + kb + half * 16);
        bf[1].h[1] = *(const v8bf*)(wrow1 + kb + half * 16 + 8);
        __builtin_prefetch(wrow0 + kb + 256, 0, 1);
#pragma unroll
        for (int mi = 0; mi < 4; ++mi) {
            // A fragment (16x32): lane ln = row m, elems e -> k = half*16 + e
            const __bf16* arow = A + (size_t)(mBase + mi * 16 + ln) * K + kb;
            FragAB af;
            af.h[0] = *(const v8bf*)(arow + half * 16);
            af.h[1] = *(const v8bf*)(arow + half * 16 + 8);
            acc[mi][0] = wmma_bf16(af, bf[0], acc[mi][0]);
            acc[mi][1] = wmma_bf16(af, bf[1], acc[mi][1]);
        }
    }

#pragma unroll
    for (int ni = 0; ni < 2; ++ni) {
        const int col = nBase + ni * 16 + ln;
        const float bv = bias[col];
#pragma unroll
        for (int mi = 0; mi < 4; ++mi) {
#pragma unroll
            for (int r = 0; r < 8; ++r) {
                const int row = mBase + mi * 16 + r + half * 8;
                const size_t oi = (size_t)row * N + col;
                float v = acc[mi][ni][r] + bv;
                if (EPI == 0) {
                    ((float*)outp)[oi] = v;
                } else if (EPI == 1) {
                    ((float*)outp)[oi] = v + residual[oi];
                } else if (EPI == 2) {
                    ((__bf16*)outp)[oi] = (__bf16)v;
                } else {
                    float gl = 0.5f * v * (1.f + erff(v * 0.70710678118654752f));
                    ((__bf16*)outp)[oi] = (__bf16)gl;
                }
            }
        }
    }
}

// ---------------------------------------------------------------------------
// Split fused QKV [M, 3*DIM] (bf16) into Q,K: [B,H,N,D]  and V^T: [B,H,D,N]
// ---------------------------------------------------------------------------
__global__ void split_qkv_kernel(const __bf16* __restrict__ qkv,
                                 __bf16* __restrict__ Q, __bf16* __restrict__ Kk,
                                 __bf16* __restrict__ Vt) {
    size_t idx = (size_t)blockIdx.x * 256 + threadIdx.x;   // < MTOK*3*DIM
    int m = (int)(idx / (3 * DIM));
    int c = (int)(idx % (3 * DIM));
    int which = c >> 10;            // 0:q 1:k 2:v
    int cc = c & (DIM - 1);
    int h = cc >> 6;
    int d = cc & (HDIM - 1);
    int b = m >> 11;
    int q = m & (SEQ - 1);
    size_t bh = (size_t)b * NHEADS + h;
    __bf16 v = qkv[idx];
    if (which == 0)      Q [(bh * SEQ + q) * HDIM + d] = v;
    else if (which == 1) Kk[(bh * SEQ + q) * HDIM + d] = v;
    else                 Vt[(bh * HDIM + d) * SEQ + q] = v;
}

// ---------------------------------------------------------------------------
// Flash attention. Grid (SEQ/64, NHEADS, BATCH), block 128 (4 waves).
// Each wave: 16 query rows, streams 64-key tiles:
//   S(16x64) = (Q*scale)(16x64) Kt   : 8 wmma   (scale folded into Q, exact)
//   online softmax: row max via v_permlane16_b32 VALU butterfly (no LDS pipe),
//                   row SUM on the matrix unit: lacc += P @ ones(32x16)
//   O(16x64)  += P(16x64) V(64x64)   : 8 wmma (+2 for lacc)
// ctx written as [B, N, H*D] bf16 (ready to be GEMM A-matrix).
// ---------------------------------------------------------------------------
__global__ __launch_bounds__(128)
void attn_kernel(const __bf16* __restrict__ Q, const __bf16* __restrict__ Kk,
                 const __bf16* __restrict__ Vt, __bf16* __restrict__ ctx) {
    __shared__ __bf16 pbuf[4][16 * 64];
    const int lane = threadIdx.x & 31;
    const int wave = threadIdx.x >> 5;
    const int ln   = lane & 15;
    const int half = lane >> 4;
    const int b = blockIdx.z, h = blockIdx.y;
    const int qBase = blockIdx.x * 64 + wave * 16;
    const size_t bh = (size_t)b * NHEADS + h;
    const __bf16* Qp = Q  + bh * (SEQ * HDIM);
    const __bf16* Kp = Kk + bh * (SEQ * HDIM);
    const __bf16* Vp = Vt + bh * (HDIM * SEQ);
    __bf16* lds = &pbuf[wave][0];

    // Q fragments, pre-scaled by 1/sqrt(64)=0.125 (power of two: exact in bf16)
    FragAB qa[2];
#pragma unroll
    for (int ks = 0; ks < 2; ++ks) {
        const __bf16* qr = Qp + (size_t)(qBase + ln) * HDIM + ks * 32;
        qa[ks].h[0] = *(const v8bf*)(qr + half * 16);
        qa[ks].h[1] = *(const v8bf*)(qr + half * 16 + 8);
#pragma unroll
        for (int e = 0; e < 16; ++e)
            qa[ks].v[e] = (__bf16)((float)qa[ks].v[e] * 0.125f);
    }

    // all-ones B fragment for row-sum-by-WMMA
    FragAB ones;
#pragma unroll
    for (int e = 0; e < 16; ++e) ones.v[e] = (__bf16)1.0f;

    float mi_[8];
    v8f lacc = {};
    v8f oacc[4] = {};
#pragma unroll
    for (int r = 0; r < 8; ++r) mi_[r] = -1e30f;

    for (int j = 0; j < SEQ; j += 64) {
        // S = Q @ K^T for 64 keys: 4 column tiles x 2 head-dim k-steps
        v8f s[4] = {};
#pragma unroll
        for (int sbl = 0; sbl < 4; ++sbl) {
#pragma unroll
            for (int ks = 0; ks < 2; ++ks) {
                const __bf16* kr = Kp + (size_t)(j + sbl * 16 + ln) * HDIM
                                      + ks * 32 + half * 16;
                FragAB kf;
                kf.h[0] = *(const v8bf*)(kr);
                kf.h[1] = *(const v8bf*)(kr + 8);
                s[sbl] = wmma_bf16(qa[ks], kf, s[sbl]);
            }
        }
        // row max: local max of 4 tiles, then 4-step VALU butterfly per row
        float rm[8];
#pragma unroll
        for (int r = 0; r < 8; ++r)
            rm[r] = fmaxf(fmaxf(s[0][r], s[1][r]), fmaxf(s[2][r], s[3][r]));
#pragma unroll
        for (int r = 0; r < 8; ++r)
            rm[r] = fmaxf(rm[r], lane16_xor(rm[r], 0x67452301u, 0xEFCDAB89u)); // ^1
#pragma unroll
        for (int r = 0; r < 8; ++r)
            rm[r] = fmaxf(rm[r], lane16_xor(rm[r], 0x54761032u, 0xDCFE98BAu)); // ^2
#pragma unroll
        for (int r = 0; r < 8; ++r)
            rm[r] = fmaxf(rm[r], lane16_xor(rm[r], 0x32107654u, 0xBA98FEDCu)); // ^4
#pragma unroll
        for (int r = 0; r < 8; ++r)
            rm[r] = fmaxf(rm[r], lane16_xor(rm[r], 0xFEDCBA98u, 0x76543210u)); // ^8

#pragma unroll
        for (int r = 0; r < 8; ++r) {
            float mnew  = fmaxf(mi_[r], rm[r]);
            float alpha = __expf(mi_[r] - mnew);
            mi_[r] = mnew;
            const int row = r + half * 8;
#pragma unroll
            for (int sbl = 0; sbl < 4; ++sbl)
                lds[row * 64 + sbl * 16 + ln] = (__bf16)__expf(s[sbl][r] - mnew);
            lacc[r] *= alpha;
#pragma unroll
            for (int t = 0; t < 4; ++t) oacc[t][r] *= alpha;
        }
        asm volatile("s_wait_dscnt 0" ::: "memory");
        // P @ [ones | V] in two 32-key k-steps
#pragma unroll
        for (int kk = 0; kk < 2; ++kk) {
            FragAB pa;   // P as A-layout fragment (16 rows x 32 keys)
            pa.h[0] = *(const v8bf*)(lds + ln * 64 + kk * 32 + half * 16);
            pa.h[1] = *(const v8bf*)(lds + ln * 64 + kk * 32 + half * 16 + 8);
            lacc = wmma_bf16(pa, ones, lacc);
#pragma unroll
            for (int t = 0; t < 4; ++t) {
                const __bf16* vr = Vp + (size_t)(t * 16 + ln) * SEQ
                                      + j + kk * 32 + half * 16;
                FragAB vf;
                vf.h[0] = *(const v8bf*)(vr);
                vf.h[1] = *(const v8bf*)(vr + 8);
                oacc[t] = wmma_bf16(pa, vf, oacc[t]);
            }
        }
    }

    // normalize and write ctx[b, q, h*64 + d]
#pragma unroll
    for (int r = 0; r < 8; ++r) {
        const float inv = 1.f / lacc[r];
        const int q = qBase + r + half * 8;
        const size_t base = ((size_t)b * SEQ + q) * DIM + h * HDIM;
#pragma unroll
        for (int t = 0; t < 4; ++t)
            ctx[base + t * 16 + ln] = (__bf16)(oacc[t][r] * inv);
    }
}

// ---------------------------------------------------------------------------
// Host-side orchestration
// ---------------------------------------------------------------------------
extern "C" void kernel_launch(void* const* d_in, const int* in_sizes, int n_in,
                              void* d_out, int out_size, void* d_ws, size_t ws_size,
                              hipStream_t stream) {
    (void)in_sizes; (void)n_in; (void)out_size; (void)ws_size;
    const float* x      = (const float*)d_in[0];
    const float* ln1_g  = (const float*)d_in[1];
    const float* ln1_b  = (const float*)d_in[2];
    const float* qkv_w  = (const float*)d_in[3];
    const float* qkv_b  = (const float*)d_in[4];
    const float* proj_w = (const float*)d_in[5];
    const float* proj_b = (const float*)d_in[6];
    const float* ln2_g  = (const float*)d_in[7];
    const float* ln2_b  = (const float*)d_in[8];
    const float* fc1_w  = (const float*)d_in[9];
    const float* fc1_b  = (const float*)d_in[10];
    const float* fc2_w  = (const float*)d_in[11];
    const float* fc2_b  = (const float*)d_in[12];
    float* out = (float*)d_out;

    char* ws = (char*)d_ws;
    size_t off = 0;
    auto alloc = [&](size_t bytes) -> char* {
        char* p = ws + off;
        off = (off + bytes + 255) & ~(size_t)255;
        return p;
    };
    __bf16* qkvw_b  = (__bf16*)alloc((size_t)3 * DIM * DIM * 2);
    __bf16* projw_b = (__bf16*)alloc((size_t)DIM * DIM * 2);
    __bf16* fc1w_b  = (__bf16*)alloc((size_t)HIDDEN * DIM * 2);
    __bf16* fc2w_b  = (__bf16*)alloc((size_t)DIM * HIDDEN * 2);
    __bf16* hbuf    = (__bf16*)alloc((size_t)MTOK * DIM * 2);      // h1, then h2
    __bf16* big     = (__bf16*)alloc((size_t)MTOK * HIDDEN * 2);   // qkv, then fc1 act
    __bf16* Qb      = (__bf16*)alloc((size_t)MTOK * DIM * 2);
    __bf16* Kb      = (__bf16*)alloc((size_t)MTOK * DIM * 2);
    __bf16* Vtb     = (__bf16*)alloc((size_t)MTOK * DIM * 2);
    __bf16* ctxb    = (__bf16*)alloc((size_t)MTOK * DIM * 2);
    float*  x1      = (float*)alloc((size_t)MTOK * DIM * 4);

    // weights -> bf16
    cvt_bf16_kernel<<<(3 * DIM * DIM + 255) / 256, 256, 0, stream>>>(qkv_w, qkvw_b, 3 * DIM * DIM);
    cvt_bf16_kernel<<<(DIM * DIM + 255) / 256, 256, 0, stream>>>(proj_w, projw_b, DIM * DIM);
    cvt_bf16_kernel<<<(HIDDEN * DIM + 255) / 256, 256, 0, stream>>>(fc1_w, fc1w_b, HIDDEN * DIM);
    cvt_bf16_kernel<<<(DIM * HIDDEN + 255) / 256, 256, 0, stream>>>(fc2_w, fc2w_b, DIM * HIDDEN);

    // h1 = LN1(x)
    ln_kernel<<<MTOK, 256, 0, stream>>>(x, ln1_g, ln1_b, hbuf);
    // qkv = h1 @ qkv_w^T + qkv_b  (bf16 out)
    gemm_kernel<2><<<dim3(3 * DIM / 64, MTOK / 128), 128, 0, stream>>>(
        hbuf, qkvw_b, qkv_b, nullptr, big, MTOK, 3 * DIM, DIM);
    // split into Q, K, V^T
    split_qkv_kernel<<<(MTOK * 3 * DIM) / 256, 256, 0, stream>>>(big, Qb, Kb, Vtb);
    // flash attention -> ctx [B, N, C]
    attn_kernel<<<dim3(SEQ / 64, NHEADS, BATCH), 128, 0, stream>>>(Qb, Kb, Vtb, ctxb);
    // x1 = x + ctx @ proj_w^T + proj_b  (fp32)
    gemm_kernel<1><<<dim3(DIM / 64, MTOK / 128), 128, 0, stream>>>(
        ctxb, projw_b, proj_b, x, x1, MTOK, DIM, DIM);
    // h2 = LN2(x1)
    ln_kernel<<<MTOK, 256, 0, stream>>>(x1, ln2_g, ln2_b, hbuf);
    // a1 = gelu(h2 @ fc1_w^T + fc1_b)  (bf16)
    gemm_kernel<3><<<dim3(HIDDEN / 64, MTOK / 128), 128, 0, stream>>>(
        hbuf, fc1w_b, fc1_b, nullptr, big, MTOK, HIDDEN, DIM);
    // out = x1 + a1 @ fc2_w^T + fc2_b  (fp32)
    gemm_kernel<1><<<dim3(DIM / 64, MTOK / 128), 128, 0, stream>>>(
        big, fc2w_b, fc2_b, x1, out, MTOK, DIM, HIDDEN);
}
